// RelativeCrossAttention_86096914416089
// MI455X (gfx1250) — matrix operand order repaired and checked by reading, not compiled, not run on hardware
//
#include <hip/hip_runtime.h>
#include <cstdint>

// ---------------------------------------------------------------------------
// Types for CDNA5 WMMA (wave32). v_wmma_f32_16x16x32_bf16: A,B = v16bf, C/D = v8f
// ---------------------------------------------------------------------------
typedef __bf16 bf16_t;
typedef bf16_t v16bf __attribute__((ext_vector_type(16)));
typedef bf16_t v8bf  __attribute__((ext_vector_type(8)));
typedef float  v8f   __attribute__((ext_vector_type(8)));
typedef float  v4f   __attribute__((ext_vector_type(4)));

__device__ __forceinline__ v8f wmma_bf16(v16bf a, v16bf b, v8f c) {
    // (neg_a, A, neg_b, B, c_mod, C, reuse_a, reuse_b)
    return __builtin_amdgcn_wmma_f32_16x16x32_bf16(false, a, false, b, (short)0, c,
                                                   false, false);
}

// A-fragment (16x32 bf16): lane holds row M=lane%16.
// Elements: [k0 .. k0+8) and [k0+16 .. k0+24), k0 = 32*kc + 8*(lane>=16).
__device__ __forceinline__ v16bf load_a_bf16(const bf16_t* row, int k0) {
    v8bf h0 = *(const v8bf*)(row + k0);
    v8bf h1 = *(const v8bf*)(row + k0 + 16);
    v16bf r;
#pragma unroll
    for (int i = 0; i < 8; ++i) { r[i] = h0[i]; r[8 + i] = h1[i]; }
    return r;
}

// Same A-fragment but source is fp32 (convert in-register to bf16).
__device__ __forceinline__ v16bf load_a_f32(const float* row, int k0) {
    v4f f0 = *(const v4f*)(row + k0);
    v4f f1 = *(const v4f*)(row + k0 + 4);
    v4f f2 = *(const v4f*)(row + k0 + 16);
    v4f f3 = *(const v4f*)(row + k0 + 20);
    v16bf r;
#pragma unroll
    for (int i = 0; i < 4; ++i) {
        r[i]      = (bf16_t)f0[i];
        r[4 + i]  = (bf16_t)f1[i];
        r[8 + i]  = (bf16_t)f2[i];
        r[12 + i] = (bf16_t)f3[i];
    }
    return r;
}

// B-fragment (32x16 bf16): lane holds column N=lane%16, K values
// [off .. off+8) and [off+16 .. off+24), off = 32*kc + 8*(lane>=16). p points at off.
__device__ __forceinline__ v16bf load_b_bf16(const bf16_t* p) {
    v8bf h0 = *(const v8bf*)(p);
    v8bf h1 = *(const v8bf*)(p + 16);
    v16bf r;
#pragma unroll
    for (int i = 0; i < 8; ++i) { r[i] = h0[i]; r[8 + i] = h1[i]; }
    return r;
}

// ---------------------------------------------------------------------------
// fp32 -> bf16 conversion kernel (weights / rel tables)
// ---------------------------------------------------------------------------
__global__ void __launch_bounds__(256)
cvt_bf16_kernel(const float* __restrict__ s, bf16_t* __restrict__ d, int n) {
    int i = blockIdx.x * 256 + threadIdx.x;
    if (i < n) d[i] = (bf16_t)s[i];
}

// ---------------------------------------------------------------------------
// Tiled GEMM:  out[m,n] = sum_k A[m,k] * W[n,k] + bias[n]   (W row-major [N,K])
// N = K = 512. Block = 256 threads (8 waves); each wave owns a 16x64 tile.
// A_F32: A is fp32 (converted in-register), else bf16.  OUT_BF16: bf16 vs fp32 out.
// Weight rows are K-contiguous -> direct 32B B-fragment loads (L2/WGP$-resident).
// A-fragment for kc+1 prefetched across the 4-WMMA group.
// ---------------------------------------------------------------------------
template <bool A_F32, bool OUT_BF16>
__global__ void __launch_bounds__(256)
gemm_bias_kernel(const void* __restrict__ Ap,
                 const bf16_t* __restrict__ W,
                 const float* __restrict__ bias,
                 int M, void* __restrict__ outp) {
    const int N = 512, K = 512;
    const int tid = threadIdx.x, wave = tid >> 5, lane = tid & 31;
    const int hi = lane >> 4, ln = lane & 15;
    const int m0 = blockIdx.x * 128 + wave * 16;
    const int n0 = blockIdx.y * 64;

    int mrow = m0 + ln;
    if (mrow >= M) mrow = M - 1;                 // clamp reads, guard stores

    const float*  arow_f = (const float*)Ap + (size_t)mrow * K;
    const bf16_t* arow_b = (const bf16_t*)Ap + (size_t)mrow * K;
    const bf16_t* wbase  = W + (size_t)(n0 + ln) * K + hi * 8;

    auto load_a = [&](int kc) -> v16bf {
        if constexpr (A_F32) return load_a_f32(arow_f + kc * 32, hi * 8);
        else                 return load_a_bf16(arow_b + kc * 32, hi * 8);
    };

    v8f acc[4] = {};
    v16bf a_cur = load_a(0);
#pragma unroll 4
    for (int kc = 0; kc < 16; ++kc) {
        v16bf a_nxt = a_cur;
        if (kc < 15) a_nxt = load_a(kc + 1);     // prefetch next K-chunk
#pragma unroll
        for (int nt = 0; nt < 4; ++nt) {
            const bf16_t* wrow = wbase + (size_t)(nt * 16) * K + kc * 32;
            acc[nt] = wmma_bf16(a_cur, load_b_bf16(wrow), acc[nt]);
        }
        a_cur = a_nxt;
    }

#pragma unroll
    for (int nt = 0; nt < 4; ++nt) {
        const int n  = n0 + nt * 16 + ln;
        const float bval = bias[n];
#pragma unroll
        for (int r = 0; r < 8; ++r) {
            const int m = m0 + r + 8 * hi;
            if (m < M) {
                const float v = acc[nt][r] + bval;
                if constexpr (OUT_BF16)
                    ((bf16_t*)outp)[(size_t)m * N + n] = (bf16_t)v;
                else
                    ((float*)outp)[(size_t)m * N + n] = v;
            }
        }
    }
}

// ---------------------------------------------------------------------------
// Attention kernel. Grid: (Lq/128, B*H). Block = 256 (8 waves, 16 q-rows each).
// LDS layout (dynamic, 256KB):
//   sK  [96][64]  bf16   (keys, rows 77..95 zero)
//   sRK [160][64] bf16   (rel_k table, rows 155..159 zero)
//   sVt [64][96]  bf16   (values, transposed: [d][k])
//   sRVt[64][160] bf16   (rel_v table, transposed: [d][r])
//   per wave: S1[16][96] f32, S2[16][160] f32, Pb[16][96] bf16, Cb[16][160] bf16
// ---------------------------------------------------------------------------
#define ATTN_SHARED_BASE (96*64*2 + 160*64*2 + 64*96*2 + 64*160*2)   /* 65536 */
#define ATTN_WAVE_BYTES  (16*96*4 + 16*160*4 + 16*96*2 + 16*160*2)   /* 24576 */
#define ATTN_LDS_BYTES   (ATTN_SHARED_BASE + 8 * ATTN_WAVE_BYTES)    /* 262144 */

__global__ void __launch_bounds__(256)
relattn_kernel(const bf16_t* __restrict__ qb,
               const bf16_t* __restrict__ kb,
               const bf16_t* __restrict__ vb,
               const bf16_t* __restrict__ rk,
               const bf16_t* __restrict__ rv,
               bf16_t* __restrict__ ao) {
    extern __shared__ char smem[];
    bf16_t* sK   = (bf16_t*)smem;                // [96][64]
    bf16_t* sRK  = sK  + 96 * 64;                // [160][64]
    bf16_t* sVt  = sRK + 160 * 64;               // [64][96]
    bf16_t* sRVt = sVt + 64 * 96;                // [64][160]

    const int tid = threadIdx.x, wave = tid >> 5, lane = tid & 31;
    const int hi = lane >> 4, ln = lane & 15;
    char* wbase = smem + ATTN_SHARED_BASE + wave * ATTN_WAVE_BYTES;
    float*  S1 = (float*)wbase;                  // [16][96]
    float*  S2 = S1 + 16 * 96;                   // [16][160]
    bf16_t* Pb = (bf16_t*)(S2 + 16 * 160);       // [16][96]
    bf16_t* Cb = Pb + 16 * 96;                   // [16][160]

    const int bh = blockIdx.y;
    const int b = bh >> 3, h = bh & 7;
    const int q0 = blockIdx.x * 128 + wave * 16;
    const int C = 512, Lk = 77, Lq = 2048;

    // ---- stage 1: cooperative LDS fill -----------------------------------
    const bf16_t* kbase = kb + (size_t)(b * Lk) * C + h * 64;
    const bf16_t* vbase = vb + (size_t)(b * Lk) * C + h * 64;
    for (int i = tid; i < 96 * 64; i += 256) {
        const int r = i >> 6, c = i & 63;
        const bool ok = (r < Lk);
        sK[r * 64 + c]  = ok ? kbase[r * C + c] : (bf16_t)0.f;
        sVt[c * 96 + r] = ok ? vbase[r * C + c] : (bf16_t)0.f;
    }
    for (int i = tid; i < 160 * 64; i += 256) {
        const int r = i >> 6, c = i & 63;
        const bool ok = (r < 155);
        sRK[r * 64 + c]   = ok ? rk[r * 64 + c] : (bf16_t)0.f;
        sRVt[c * 160 + r] = ok ? rv[r * 64 + c] : (bf16_t)0.f;
    }
    __syncthreads();

    // ---- stage 2: scores S1 = Q K^T, S2 = Q relK^T (B-frags double-buffered)
    const bf16_t* qrow = qb + (size_t)(b * Lq + q0 + ln) * C + h * 64;
    const v16bf aq0 = load_a_bf16(qrow, 0 * 32 + hi * 8);
    const v16bf aq1 = load_a_bf16(qrow, 1 * 32 + hi * 8);

    {
        const bf16_t* base = sK + ln * 64 + hi * 8;
        v16bf b0 = load_b_bf16(base);
        v16bf b1 = load_b_bf16(base + 32);
#pragma unroll
        for (int nt = 0; nt < 6; ++nt) {
            v16bf p0 = b0, p1 = b1;
            if (nt < 5) {                         // prefetch next tile
                const bf16_t* nb = base + (nt + 1) * 16 * 64;
                p0 = load_b_bf16(nb);
                p1 = load_b_bf16(nb + 32);
            }
            v8f acc = {};
            acc = wmma_bf16(aq0, b0, acc);
            acc = wmma_bf16(aq1, b1, acc);
#pragma unroll
            for (int r = 0; r < 8; ++r)
                S1[(r + 8 * hi) * 96 + nt * 16 + ln] = acc[r];
            b0 = p0; b1 = p1;
        }
    }
    {
        const bf16_t* base = sRK + ln * 64 + hi * 8;
        v16bf b0 = load_b_bf16(base);
        v16bf b1 = load_b_bf16(base + 32);
#pragma unroll
        for (int nt = 0; nt < 10; ++nt) {
            v16bf p0 = b0, p1 = b1;
            if (nt < 9) {
                const bf16_t* nb = base + (nt + 1) * 16 * 64;
                p0 = load_b_bf16(nb);
                p1 = load_b_bf16(nb + 32);
            }
            v8f acc = {};
            acc = wmma_bf16(aq0, b0, acc);
            acc = wmma_bf16(aq1, b1, acc);
#pragma unroll
            for (int r = 0; r < 8; ++r)
                S2[(r + 8 * hi) * 160 + nt * 16 + ln] = acc[r];
            b0 = p0; b1 = p1;
        }
    }
    __syncthreads();

    // ---- stage 3: softmax + gather(S2) + scatter(Cb); lane pair (ln, ln+16)
    // shares row ln: lo lane keys [0,39), hi lane keys [39,77).
    {
        const int row = ln;
        const int qg  = q0 + row;
        const float scale = 0.044194173824159216f;   // 1/sqrt(512)
        float* s1r = S1 + row * 96;
        float* s2r = S2 + row * 160;
        bf16_t* pr = Pb + row * 96;
        bf16_t* cr = Cb + row * 160;
        const int k0 = hi ? 39 : 0;
        const int k1 = hi ? 77 : 39;

        float m = -1e30f, ssum = 0.f;
        for (int k = k0; k < k1; ++k) {
            int d = k - qg; d = d < -77 ? -77 : (d > 77 ? 77 : d);
            const float xv = (s1r[k] + s2r[d + 77]) * scale;
            const float mn = fmaxf(m, xv);
            ssum = ssum * __expf(m - mn) + __expf(xv - mn);
            m = mn;
        }
        // combine partial (max, sum) across the lane pair
        const float mo = __shfl_xor(m, 16, 32);
        const float so = __shfl_xor(ssum, 16, 32);
        const float mt = fmaxf(m, mo);
        const float st = ssum * __expf(m - mt) + so * __expf(mo - mt);
        const float inv = 1.f / st;

        // zero-fill (in-order LDS within wave guarantees visibility before scatter)
        if (!hi)
            for (int k = Lk; k < 96; ++k) pr[k] = (bf16_t)0.f;
        for (int r = hi * 80; r < hi * 80 + 80; ++r) cr[r] = (bf16_t)0.f;

        // probs + scatter. idx is strictly increasing except the idx==0 clamp run,
        // so every idx>0 is written by exactly one lane; idx==0 is pair-combined.
        float zero_sum = 0.f;
        int cur = -1; float runv = 0.f;
        for (int k = k0; k < k1; ++k) {
            int d = k - qg; d = d < -77 ? -77 : (d > 77 ? 77 : d);
            const int id = d + 77;
            const float p = __expf((s1r[k] + s2r[id]) * scale - mt) * inv;
            pr[k] = (bf16_t)p;
            if (id == 0)          { zero_sum += p; }
            else if (id == cur)   { runv += p; }
            else { if (cur > 0) cr[cur] = (bf16_t)runv; cur = id; runv = p; }
        }
        if (cur > 0) cr[cur] = (bf16_t)runv;
        const float zo = __shfl_xor(zero_sum, 16, 32);
        if (!hi) cr[0] = (bf16_t)(zero_sum + zo);
    }
    __syncthreads();

    // ---- stage 4: out = Pb @ V + Cb @ relV; A-frags hoisted (nt-invariant) --
    v16bf ap[3], ac[5];
#pragma unroll
    for (int kc = 0; kc < 3; ++kc) ap[kc] = load_a_bf16(Pb + ln * 96,  kc * 32 + hi * 8);
#pragma unroll
    for (int kc = 0; kc < 5; ++kc) ac[kc] = load_a_bf16(Cb + ln * 160, kc * 32 + hi * 8);

#pragma unroll
    for (int nt = 0; nt < 4; ++nt) {
        v8f acc = {};
        const bf16_t* vrow  = sVt  + (nt * 16 + ln) * 96  + hi * 8;
        const bf16_t* rvrow = sRVt + (nt * 16 + ln) * 160 + hi * 8;
#pragma unroll
        for (int kc = 0; kc < 3; ++kc)
            acc = wmma_bf16(ap[kc], load_b_bf16(vrow + kc * 32), acc);
#pragma unroll
        for (int kc = 0; kc < 5; ++kc)
            acc = wmma_bf16(ac[kc], load_b_bf16(rvrow + kc * 32), acc);

        const int col = h * 64 + nt * 16 + ln;
#pragma unroll
        for (int r = 0; r < 8; ++r) {
            const int m = q0 + r + 8 * hi;
            ao[(size_t)(b * Lq + m) * C + col] = (bf16_t)acc[r];
        }
    }
}

// ---------------------------------------------------------------------------
// Host launcher
// ---------------------------------------------------------------------------
extern "C" void kernel_launch(void* const* d_in, const int* in_sizes, int n_in,
                              void* d_out, int out_size, void* d_ws, size_t ws_size,
                              hipStream_t stream) {
    (void)in_sizes; (void)n_in; (void)out_size; (void)ws_size;
    const float* x    = (const float*)d_in[0];   // [32,2048,512]
    const float* cond = (const float*)d_in[1];   // [32,77,512]
    const float* Wq   = (const float*)d_in[2];
    const float* bq   = (const float*)d_in[3];
    const float* Wk   = (const float*)d_in[4];
    const float* bk   = (const float*)d_in[5];
    const float* Wv   = (const float*)d_in[6];
    const float* bv   = (const float*)d_in[7];
    const float* Wp   = (const float*)d_in[8];
    const float* bp   = (const float*)d_in[9];
    const float* relk = (const float*)d_in[10];  // [155,64]
    const float* relv = (const float*)d_in[11];

    const int B = 32, Lq = 2048, Lk = 77, C = 512;
    const int Mq = B * Lq;          // 65536
    const int Mk = B * Lk;          // 2464

    // workspace carve-up (bytes), all 16B aligned
    char* ws = (char*)d_ws;
    size_t o = 0;
    bf16_t* Wq_b = (bf16_t*)(ws + o); o += (size_t)C * C * 2;
    bf16_t* Wk_b = (bf16_t*)(ws + o); o += (size_t)C * C * 2;
    bf16_t* Wv_b = (bf16_t*)(ws + o); o += (size_t)C * C * 2;
    bf16_t* Wp_b = (bf16_t*)(ws + o); o += (size_t)C * C * 2;
    bf16_t* rk_b = (bf16_t*)(ws + o); o += (size_t)155 * 64 * 2;
    bf16_t* rv_b = (bf16_t*)(ws + o); o += (size_t)155 * 64 * 2;
    bf16_t* q_b  = (bf16_t*)(ws + o); o += (size_t)Mq * C * 2;
    bf16_t* k_b  = (bf16_t*)(ws + o); o += (size_t)Mk * C * 2;
    bf16_t* v_b  = (bf16_t*)(ws + o); o += (size_t)Mk * C * 2;
    bf16_t* ao_b = (bf16_t*)(ws + o); o += (size_t)Mq * C * 2;

    // 1) convert weights & rel tables to bf16
    const int nW = C * C, nR = 155 * 64;
    cvt_bf16_kernel<<<(nW + 255) / 256, 256, 0, stream>>>(Wq, Wq_b, nW);
    cvt_bf16_kernel<<<(nW + 255) / 256, 256, 0, stream>>>(Wk, Wk_b, nW);
    cvt_bf16_kernel<<<(nW + 255) / 256, 256, 0, stream>>>(Wv, Wv_b, nW);
    cvt_bf16_kernel<<<(nW + 255) / 256, 256, 0, stream>>>(Wp, Wp_b, nW);
    cvt_bf16_kernel<<<(nR + 255) / 256, 256, 0, stream>>>(relk, rk_b, nR);
    cvt_bf16_kernel<<<(nR + 255) / 256, 256, 0, stream>>>(relv, rv_b, nR);

    // 2) projections (bf16 WMMA, fp32 A converted in-register, bf16 output)
    {
        dim3 grid((Mq + 127) / 128, C / 64);
        gemm_bias_kernel<true, true><<<grid, 256, 0, stream>>>(x, Wq_b, bq, Mq, q_b);
    }
    {
        dim3 grid((Mk + 127) / 128, C / 64);
        gemm_bias_kernel<true, true><<<grid, 256, 0, stream>>>(cond, Wk_b, bk, Mk, k_b);
        gemm_bias_kernel<true, true><<<grid, 256, 0, stream>>>(cond, Wv_b, bv, Mk, v_b);
    }

    // 3) attention (flash-style, all-LDS, bf16 WMMA)
    {
        static_assert(ATTN_LDS_BYTES == 262144, "lds layout");
        hipFuncSetAttribute(reinterpret_cast<const void*>(&relattn_kernel),
                            hipFuncAttributeMaxDynamicSharedMemorySize, ATTN_LDS_BYTES);
        dim3 grid(Lq / 128, B * 8);
        relattn_kernel<<<grid, 256, ATTN_LDS_BYTES, stream>>>(q_b, k_b, v_b, rk_b, rv_b, ao_b);
    }

    // 4) output projection (bf16 A, fp32 out + bias)
    {
        dim3 grid((Mq + 127) / 128, C / 64);
        gemm_bias_kernel<false, false><<<grid, 256, 0, stream>>>(ao_b, Wp_b, bp, Mq, d_out);
    }
}
